// GINEncoder_23837068492865
// MI455X (gfx1250) — matrix operand order repaired and checked by reading, not compiled, not run on hardware
//
#include <hip/hip_runtime.h>
#include <hip/hip_bf16.h>

typedef __attribute__((ext_vector_type(16))) __bf16 v16bf;
typedef __attribute__((ext_vector_type(8)))  float  v8f;

#define H 64
#define BN_EPS 1e-5f

// ---------------- utility kernels ----------------

__global__ void zero_f32(float* __restrict__ p, int n) {
    int t = blockIdx.x * blockDim.x + threadIdx.x;
    if (t < n) p[t] = 0.f;
}

__global__ void add2_f32(const float* __restrict__ a, const float* __restrict__ b,
                         float* __restrict__ o, int n) {
    int t = blockIdx.x * blockDim.x + threadIdx.x;
    if (t < n) o[t] = a[t] + b[t];
}

// ---------------- weight swizzle: f32 [64,64] -> bf16 in WMMA B-fragment order ----
// Reader (lane = half*16+m, fragment (kt,ct), element i) wants W[(kt*32+half*8+i%8+16*(i/8))*64 + ct*16 + m].
// Store at ((kt*4+ct)*32 + half*16 + m)*16 + i.  krows<64 => zero-pad rows (layer-0 W1).
__global__ void swizzle_w_bf16(const float* __restrict__ W, int krows,
                               __bf16* __restrict__ out) {
    int t = blockIdx.x * blockDim.x + threadIdx.x;   // 4096
    if (t >= H * H) return;
    int k = t >> 6, c = t & (H - 1);
    float v = (k < krows) ? W[k * H + c] : 0.f;
    int kt = k >> 5, kk = k & 31;
    int half = (kk >> 3) & 1;      // B-fragment mapping: i<8 -> kk = half*8 + i; i>=8 -> kk = 16 + half*8 + (i-8)
    int i    = (kk & 7) + ((kk & 16) ? 8 : 0);
    int m = c & 15, ct = c >> 4;
    out[(((kt * 4 + ct) * 32) + half * 16 + m) * 16 + i] = (__bf16)v;
}

// ---------------- layer-0 helpers (IN_DIM = 3) ----------------

__global__ void scatter_add3(const float* __restrict__ x3, const int* __restrict__ src,
                             const int* __restrict__ dst, float* __restrict__ agg, int E) {
    int e = blockIdx.x * blockDim.x + threadIdx.x;
    if (e >= E) return;
    int s = src[e], d = dst[e];
    const float* xr = x3 + (size_t)s * 3;
    float* ar = agg + (size_t)d * H;
    atomicAdd(ar + 0, xr[0]);
    atomicAdd(ar + 1, xr[1]);
    atomicAdd(ar + 2, xr[2]);
}

// xin[n,64] = concat(x3 + agg[:, :3], zeros)
__global__ void build_xin0(const float* __restrict__ x3, const float* __restrict__ agg,
                           float* __restrict__ xin, int total) {
    int t = blockIdx.x * blockDim.x + threadIdx.x;
    if (t >= total) return;
    int c = t & (H - 1);
    int r = t >> 6;
    xin[t] = (c < 3) ? (x3[(size_t)r * 3 + c] + agg[t]) : 0.f;
}

// ---------------- edge scatter, d = 64 ----------------
__global__ void scatter_add64(const float* __restrict__ x, const int* __restrict__ src,
                              const int* __restrict__ dst, float* __restrict__ agg, int E) {
    long long t = (long long)blockIdx.x * blockDim.x + threadIdx.x;
    if (t >= (long long)E * 16) return;
    int e = (int)(t >> 4);
    int q = ((int)t & 15) * 4;
    int s = src[e], d = dst[e];
    const float4 v = *(const float4*)(x + (size_t)s * H + q);
    float* p = agg + (size_t)d * H + q;
    atomicAdd(p + 0, v.x);
    atomicAdd(p + 1, v.y);
    atomicAdd(p + 2, v.z);
    atomicAdd(p + 3, v.w);
}

// ---------------- fused GIN MLP via bf16 WMMA, swizzled fragments ----------------

static __device__ __forceinline__ void pack4(v16bf& dst, int base, float4 q) {
    dst[base + 0] = (__bf16)q.x;
    dst[base + 1] = (__bf16)q.y;
    dst[base + 2] = (__bf16)q.z;
    dst[base + 3] = (__bf16)q.w;
}

// 128 threads = 4 waves; each wave owns a 16-row strip; 64 rows per block.
__global__ __launch_bounds__(128) void mlp_wmma(
    const float*  __restrict__ xin,  // [n,64]
    const __bf16* __restrict__ w1z,  // [4096] swizzled
    const float*  __restrict__ Bi1,  // [64]
    const __bf16* __restrict__ w2z,  // [4096] swizzled
    const float*  __restrict__ Bi2,  // [64]
    float* __restrict__ hout,        // [n,64]
    int n)
{
    __shared__ __align__(32) __bf16 w1s[H * H];
    __shared__ __align__(32) __bf16 w2s[H * H];
    __shared__ __align__(32) __bf16 h1s[H * H];  // GEMM2 A operand, A-fragment layout

    // stage swizzled weights: straight 16KB vector copy
    {
        const uint4* s1 = (const uint4*)w1z;
        const uint4* s2 = (const uint4*)w2z;
        uint4* d1 = (uint4*)w1s;
        uint4* d2 = (uint4*)w2s;
        for (int t = threadIdx.x; t < (H * H) / 8; t += 128) {
            d1[t] = s1[t];
            d2[t] = s2[t];
        }
    }
    __syncthreads();

    const int wave = threadIdx.x >> 5;
    const int lane = threadIdx.x & 31;
    const int half = lane >> 4;
    const int m    = lane & 15;
    const int row0 = blockIdx.x * 64 + wave * 16;
    int arow = row0 + m;
    if (arow >= n) arow = n - 1;   // clamp: keep EXEC all-ones for WMMA

    const v16bf* w1f = (const v16bf*)w1s;   // fragment-major
    const v16bf* w2f = (const v16bf*)w2s;
    const v16bf* h1f = (const v16bf*)h1s;

    // ---- A fragments for GEMM1 (global f32 -> bf16), K=64 as two K=32 tiles ----
    // a0[i]: i<8 -> xr[half*8+i]; i>=8 -> xr[16+half*8+(i-8)]; a1 same +32.
    v16bf a0, a1;
    {
        const float* xr = xin + (size_t)arow * H;
        float4 q0 = *(const float4*)(xr + half * 8);
        float4 q1 = *(const float4*)(xr + half * 8 + 4);
        float4 q2 = *(const float4*)(xr + 16 + half * 8);
        float4 q3 = *(const float4*)(xr + 16 + half * 8 + 4);
        float4 q4 = *(const float4*)(xr + 32 + half * 8);
        float4 q5 = *(const float4*)(xr + 32 + half * 8 + 4);
        float4 q6 = *(const float4*)(xr + 48 + half * 8);
        float4 q7 = *(const float4*)(xr + 48 + half * 8 + 4);
        pack4(a0, 0, q0);  pack4(a0, 4, q1);  pack4(a0, 8, q2);  pack4(a0, 12, q3);
        pack4(a1, 0, q4);  pack4(a1, 4, q5);  pack4(a1, 8, q6);  pack4(a1, 12, q7);
    }

    // ---- GEMM1: h1 = relu(xin @ W1 + b1) -> LDS (A-fragment layout) ----
#pragma unroll
    for (int ct = 0; ct < 4; ++ct) {
        v8f acc = {};
        v16bf b0 = w1f[(0 * 4 + ct) * 32 + lane];
        v16bf b1 = w1f[(1 * 4 + ct) * 32 + lane];
        acc = __builtin_amdgcn_wmma_f32_16x16x32_bf16(false, a0, false, b0,
                                                      (short)0, acc, false, false);
        acc = __builtin_amdgcn_wmma_f32_16x16x32_bf16(false, a1, false, b1,
                                                      (short)0, acc, false, false);
        // column of h1 this lane holds:
        int c  = ct * 16 + m;
        // A-layout position of k-index c within its 64-entry row:
        int kt_c = c >> 5, kk = c & 31;
        int pos  = kt_c * 32 + ((kk >> 3) & 1) * 16 + (kk & 7) + ((kk & 16) ? 8 : 0);
        float bias = Bi1[c];
#pragma unroll
        for (int v = 0; v < 8; ++v) {
            int r = wave * 16 + v + half * 8;
            h1s[r * H + pos] = (__bf16)fmaxf(acc[v] + bias, 0.f);
        }
    }
    __syncthreads();

    // ---- A fragments for GEMM2: contiguous v16bf reads from swizzled h1 ----
    int rbase = (wave * 16 + m) * 4;          // v16bf index = row*4 + kt*2 + half
    v16bf c0v = h1f[rbase + 0 * 2 + half];
    v16bf c1v = h1f[rbase + 1 * 2 + half];

    // ---- GEMM2: hout = h1 @ W2 + b2 ----
#pragma unroll
    for (int ct = 0; ct < 4; ++ct) {
        v8f acc = {};
        v16bf b0 = w2f[(0 * 4 + ct) * 32 + lane];
        v16bf b1 = w2f[(1 * 4 + ct) * 32 + lane];
        acc = __builtin_amdgcn_wmma_f32_16x16x32_bf16(false, c0v, false, b0,
                                                      (short)0, acc, false, false);
        acc = __builtin_amdgcn_wmma_f32_16x16x32_bf16(false, c1v, false, b1,
                                                      (short)0, acc, false, false);
        int c = ct * 16 + m;
        float bias = Bi2[c];
#pragma unroll
        for (int v = 0; v < 8; ++v) {
            int grow = row0 + v + half * 8;
            if (grow < n) hout[(size_t)grow * H + c] = acc[v] + bias;
        }
    }
}

// ---------------- BatchNorm (training stats over N rows) ----------------

__global__ __launch_bounds__(256) void bn_stats(const float* __restrict__ h,
                                                float* __restrict__ stats, int n) {
    int c = threadIdx.x & (H - 1);
    int g = threadIdx.x >> 6;            // 0..3 row groups
    float s = 0.f, ss = 0.f;
    for (int r = blockIdx.x * 4 + g; r < n; r += gridDim.x * 4) {
        float v = h[(size_t)r * H + c];
        s += v;
        ss += v * v;
    }
    __shared__ float red[512];
    red[threadIdx.x] = s;
    red[256 + threadIdx.x] = ss;
    __syncthreads();
    if (threadIdx.x < H) {
        float S  = red[c] + red[64 + c] + red[128 + c] + red[192 + c];
        float SS = red[256 + c] + red[256 + 64 + c] + red[256 + 128 + c] + red[256 + 192 + c];
        atomicAdd(&stats[c], S);
        atomicAdd(&stats[H + c], SS);
    }
}

__global__ void bn_finalize(const float* __restrict__ stats, const float* __restrict__ gamma,
                            const float* __restrict__ beta, float* __restrict__ bnp, float inv_n) {
    int c = threadIdx.x;
    if (c >= H) return;
    float mu  = stats[c] * inv_n;
    float var = stats[H + c] * inv_n - mu * mu;
    float sc  = gamma[c] * rsqrtf(var + BN_EPS);
    bnp[c]     = sc;
    bnp[H + c] = beta[c] - mu * sc;
}

__global__ void bn_apply_relu(const float* __restrict__ h, const float* __restrict__ bnp,
                              float* __restrict__ xo, int total) {
    int t = blockIdx.x * blockDim.x + threadIdx.x;
    if (t >= total) return;
    int c = t & (H - 1);
    xo[t] = fmaxf(h[t] * bnp[c] + bnp[H + c], 0.f);
}

// ---------------- global mean pool + output linear ----------------

__global__ void pool_accum(const float* __restrict__ x, const int* __restrict__ batch,
                           float* __restrict__ gsum, int total) {
    int t = blockIdx.x * blockDim.x + threadIdx.x;
    if (t >= total) return;
    int c = t & (H - 1);
    int r = t >> 6;
    atomicAdd(&gsum[(size_t)batch[r] * H + c], x[t]);
}

__global__ void pool_count(const int* __restrict__ batch, float* __restrict__ gcnt, int n) {
    int r = blockIdx.x * blockDim.x + threadIdx.x;
    if (r < n) atomicAdd(&gcnt[batch[r]], 1.f);
}

__global__ __launch_bounds__(64) void out_gemm(const float* __restrict__ gsum,
                                               const float* __restrict__ gcnt,
                                               const float* __restrict__ wout,
                                               const float* __restrict__ bout,
                                               float* __restrict__ out) {
    __shared__ float row[H];
    int g = blockIdx.x, j = threadIdx.x;
    float cnt = gcnt[g];
    row[j] = (cnt > 0.f) ? (gsum[(size_t)g * H + j] / cnt) : 0.f;
    __syncthreads();
    float acc = bout[j];
#pragma unroll 8
    for (int k = 0; k < H; ++k) acc += row[k] * wout[k * H + j];
    out[(size_t)g * H + j] = acc;
}

// ---------------- host side ----------------

static inline int cdiv(long long a, int b) { return (int)((a + b - 1) / b); }

extern "C" void kernel_launch(void* const* d_in, const int* in_sizes, int n_in,
                              void* d_out, int out_size, void* d_ws, size_t ws_size,
                              hipStream_t stream) {
    const float* x0    = (const float*)d_in[0];   // [N,3]
    const int*   eidx  = (const int*)d_in[1];     // [2,E]
    const int*   batch = (const int*)d_in[2];     // [N]
    const float* w1_0  = (const float*)d_in[3];   // [3,64]
    const float* w1_r  = (const float*)d_in[4];   // [2,64,64]
    const float* b1    = (const float*)d_in[5];   // [3,64]
    const float* w2    = (const float*)d_in[6];   // [3,64,64]
    const float* b2    = (const float*)d_in[7];   // [3,64]
    const float* gamma = (const float*)d_in[8];   // [3,64]
    const float* beta  = (const float*)d_in[9];   // [3,64]
    const float* wout  = (const float*)d_in[10];  // [64,64]
    const float* bout  = (const float*)d_in[11];  // [64]
    float* out = (float*)d_out;

    const int N = in_sizes[0] / 3;
    const int E = in_sizes[1] / 2;
    const int G = out_size / H;
    const int NH = N * H;

    const int* src = eidx;
    const int* dst = eidx + E;

    // workspace layout (floats)
    float* ws    = (float*)d_ws;
    float* xcur  = ws;                   // N*64
    float* agg   = xcur + (size_t)NH;    // N*64
    float* xin   = agg + (size_t)NH;     // N*64
    float* h     = xin + (size_t)NH;     // N*64
    float* stats = h + (size_t)NH;       // 128
    float* bnp   = stats + 2 * H;        // 128
    float* gsum  = bnp + 2 * H;          // G*64
    float* gcnt  = gsum + (size_t)G * H; // G
    __bf16* w1z  = (__bf16*)(gcnt + G);  // 4096 bf16 (2048 floats worth)
    __bf16* w2z  = w1z + H * H;          // 4096 bf16

    const int TB = 256;

    // ---------- layer 0 ----------
    zero_f32<<<cdiv(NH, TB), TB, 0, stream>>>(agg, NH);
    scatter_add3<<<cdiv(E, TB), TB, 0, stream>>>(x0, src, dst, agg, E);
    build_xin0<<<cdiv(NH, TB), TB, 0, stream>>>(x0, agg, xin, NH);
    swizzle_w_bf16<<<cdiv(H * H, TB), TB, 0, stream>>>(w1_0, 3, w1z);
    swizzle_w_bf16<<<cdiv(H * H, TB), TB, 0, stream>>>(w2, H, w2z);
    mlp_wmma<<<cdiv(N, 64), 128, 0, stream>>>(xin, w1z, b1, w2z, b2, h, N);
    zero_f32<<<1, TB, 0, stream>>>(stats, 2 * H);
    bn_stats<<<512, 256, 0, stream>>>(h, stats, N);
    bn_finalize<<<1, H, 0, stream>>>(stats, gamma, beta, bnp, 1.f / (float)N);
    bn_apply_relu<<<cdiv(NH, TB), TB, 0, stream>>>(h, bnp, xcur, NH);

    // ---------- layers 1..2 ----------
    for (int i = 1; i < 3; ++i) {
        zero_f32<<<cdiv(NH, TB), TB, 0, stream>>>(agg, NH);
        scatter_add64<<<cdiv((long long)E * 16, TB), TB, 0, stream>>>(xcur, src, dst, agg, E);
        add2_f32<<<cdiv(NH, TB), TB, 0, stream>>>(xcur, agg, xin, NH);
        swizzle_w_bf16<<<cdiv(H * H, TB), TB, 0, stream>>>(w1_r + (size_t)(i - 1) * H * H, H, w1z);
        swizzle_w_bf16<<<cdiv(H * H, TB), TB, 0, stream>>>(w2 + (size_t)i * H * H, H, w2z);
        mlp_wmma<<<cdiv(N, 64), 128, 0, stream>>>(xin, w1z, b1 + i * H, w2z, b2 + i * H, h, N);
        zero_f32<<<1, TB, 0, stream>>>(stats, 2 * H);
        bn_stats<<<512, 256, 0, stream>>>(h, stats, N);
        bn_finalize<<<1, H, 0, stream>>>(stats, gamma + i * H, beta + i * H, bnp, 1.f / (float)N);
        bn_apply_relu<<<cdiv(NH, TB), TB, 0, stream>>>(h, bnp, xcur, NH);
    }

    // ---------- global mean pool + output linear ----------
    zero_f32<<<cdiv(G * H + G, TB), TB, 0, stream>>>(gsum, G * H + G); // gsum + gcnt contiguous
    pool_accum<<<cdiv(NH, TB), TB, 0, stream>>>(xcur, batch, gsum, NH);
    pool_count<<<cdiv(N, TB), TB, 0, stream>>>(batch, gcnt, N);
    out_gemm<<<G, H, 0, stream>>>(gsum, gcnt, wout, bout, out);
}
